// ADTNLinear_18786186953045
// MI455X (gfx1250) — compile-verified, matrix operand order checked by reading, not run on hardware
//
#include <hip/hip_runtime.h>

typedef __attribute__((ext_vector_type(16))) __bf16  v16bf;
typedef __attribute__((ext_vector_type(8)))  __bf16  v8bf;
typedef __attribute__((ext_vector_type(2)))  __bf16  v2bf;
typedef __attribute__((ext_vector_type(8)))  float   v8f;
typedef __attribute__((ext_vector_type(4)))  float   f32x4;

#define IN_F     4096
#define OUT_F    4096
#define GROUP    64
#define N_GROUPS 64
#define N_SUB    3
#define TOK_TILE 64
#define THREADS  128

// saddr-form global access: 64-bit SGPR base + zext(32-bit VGPR byte offset).
__device__ __forceinline__ float ldg_off(const float* base, unsigned byteOff) {
  return *(const float*)((const char*)base + (unsigned long long)byteOff);
}
__device__ __forceinline__ void stg_off(float* base, unsigned byteOff, float v) {
  *(float*)((char*)base + (unsigned long long)byteOff) = v;
}

__global__ __launch_bounds__(THREADS)
void adtn_wmma_kernel(const float* __restrict__ x,
                      const float* __restrict__ W,
                      const float* __restrict__ bias,
                      const long long* __restrict__ perms,
                      float* __restrict__ out) {
  // W block staged transposed: [l][o][i] in bf16 -> B-fragment reads are
  // contiguous 16B LDS loads per lane. 24KB + 768B indices.
  __shared__ __align__(16) __bf16 s_w[N_SUB * GROUP * GROUP];
  __shared__ unsigned s_chan[N_SUB * GROUP];      // pre-scaled byte offsets

  const int g       = blockIdx.x;
  const int tokBase = blockIdx.y * TOK_TILE;
  const int tid     = threadIdx.x;

  // ---- stage permutation indices (as byte offsets) for this group ----
  for (int t = tid; t < N_SUB * GROUP; t += THREADS) {
    const int l = t >> 6, i = t & 63;
    s_chan[t] = (unsigned)perms[(long)l * IN_F + g * GROUP + i] << 2;
  }
  // ---- stage W[l][g][:][:] as bf16, transposed to [l][o][i] ----
  // o contiguous in global -> f32x4 loads; i-pairs packed -> ds_store_b32.
  for (int t = tid; t < (N_SUB * GROUP * GROUP) / 8; t += THREADS) {
    const int l  = t >> 9;            // 512 (ip,oq) pairs per sublayer
    const int r  = t & 511;
    const int ip = r >> 4;            // i = 2*ip, 2*ip+1
    const int oq = (r & 15) << 2;     // o quad base
    const long base = (long)l * (N_GROUPS * GROUP * GROUP) + g * (GROUP * GROUP);
    const f32x4 w0 = *(const f32x4*)&W[base + (2 * ip) * GROUP + oq];
    const f32x4 w1 = *(const f32x4*)&W[base + (2 * ip + 1) * GROUP + oq];
    #pragma unroll
    for (int j = 0; j < 4; ++j) {
      v2bf p;
      p[0] = (__bf16)w0[j];
      p[1] = (__bf16)w1[j];
      *(v2bf*)&s_w[(l * GROUP + oq + j) * GROUP + 2 * ip] = p;  // 4B aligned
    }
  }
  __syncthreads();

  const int wave = tid >> 5;
  const int lane = tid & 31;
  const int h    = lane >> 4;      // lane half selects K sub-block
  const int ln   = lane & 15;      // M for A-frag, N for B-frag, N for C/D
  const unsigned rowByte = (unsigned)(tokBase + wave * 16 + ln) * (IN_F * 4u);

  v8f acc[4];
  #pragma unroll
  for (int nt = 0; nt < 4; ++nt) acc[nt] = (v8f){0.f,0.f,0.f,0.f,0.f,0.f,0.f,0.f};

  #pragma unroll
  for (int l = 0; l < N_SUB; ++l) {
    #pragma unroll
    for (int kb = 0; kb < 2; ++kb) {       // K = 64 -> two 16x16x32 steps
      // ----- A fragment (16-bit A 16x32 ISA layout): gathered x loads -----
      v16bf a;
      #pragma unroll
      for (int idx = 0; idx < 16; ++idx) {
        const int v = idx >> 1, hw = idx & 1;
        const int k = ((v >> 2) << 4) + h * 8 + ((v & 3) << 1) + hw;
        const unsigned chB = s_chan[l * GROUP + kb * 32 + k];
        a[idx] = (__bf16)ldg_off(x, rowByte + chB);   // saddr global_load_b32
      }

      // ----- 4 N-tiles of the 64-wide output group -----
      #pragma unroll
      for (int nt = 0; nt < 4; ++nt) {
        const int n = nt * 16 + ln;
        const v8bf* bp =
            (const v8bf*)&s_w[(l * GROUP + n) * GROUP + kb * 32 + h * 8];
        const v8bf blo = bp[0];   // K = kb*32 + h*8 + (0..7)
        const v8bf bhi = bp[2];   // K = kb*32 + 16 + h*8 + (0..7)
        v16bf b;
        #pragma unroll
        for (int j = 0; j < 8; ++j) { b[j] = blo[j]; b[j + 8] = bhi[j]; }

        acc[nt] = __builtin_amdgcn_wmma_f32_16x16x32_bf16(
            false, a, false, b, (short)0, acc[nt], false, false);
      }
    }
  }

  // ----- bias + store (32-bit C/D 16x16 ISA layout) -----
  const unsigned outRowByte0 = (unsigned)(tokBase + wave * 16) * (OUT_F * 4u);
  #pragma unroll
  for (int nt = 0; nt < 4; ++nt) {
    const int col = g * GROUP + nt * 16 + ln;
    const float bv = bias[col];
    const unsigned colByte = (unsigned)col << 2;
    #pragma unroll
    for (int r = 0; r < 8; ++r) {
      const int m = r + 8 * h;
      stg_off(out, outRowByte0 + (unsigned)m * (OUT_F * 4u) + colByte,
              acc[nt][r] + bv);                       // saddr global_store_b32
    }
  }
}

extern "C" void kernel_launch(void* const* d_in, const int* in_sizes, int n_in,
                              void* d_out, int out_size, void* d_ws, size_t ws_size,
                              hipStream_t stream) {
  const float*     x     = (const float*)d_in[0];
  const float*     W     = (const float*)d_in[1];
  const float*     bias  = (const float*)d_in[2];
  const long long* perms = (const long long*)d_in[3];
  float*           out   = (float*)d_out;

  const int n_tokens = in_sizes[0] / IN_F;   // B*S = 16384
  dim3 grid(N_GROUPS, n_tokens / TOK_TILE);  // (64, 256)
  adtn_wmma_kernel<<<grid, THREADS, 0, stream>>>(x, W, bias, perms, out);
}
